// Naive_UNet_83193516523820
// MI455X (gfx1250) — matrix operand order repaired and checked by reading, not compiled, not run on hardware
//
#include <hip/hip_runtime.h>

// ======================= types / helpers =======================
typedef __attribute__((ext_vector_type(16))) __bf16    v16bf;
typedef __attribute__((ext_vector_type(8)))  float     v8f;
typedef __attribute__((ext_vector_type(4)))  float     v4f;
typedef __attribute__((ext_vector_type(4)))  unsigned  u32x4;
typedef __attribute__((ext_vector_type(4)))  int       i32x4;
typedef __attribute__((ext_vector_type(8)))  int       i32x8;

#define BN_EPS 1e-4f

#if defined(__has_builtin)
#  if __has_builtin(__builtin_amdgcn_tensor_load_to_lds)
#    define HAVE_TDM 1
#  endif
#endif
#ifndef HAVE_TDM
#  define HAVE_TDM 0
#endif

static __device__ __forceinline__ unsigned short f32_bf16(float f) {
  union { float f; unsigned u; } x; x.f = f;
  unsigned r = x.u + 0x7FFFu + ((x.u >> 16) & 1u);   // round-to-nearest-even
  return (unsigned short)(r >> 16);
}

union BF16Frag { v16bf v; unsigned short u[16]; u32x4 q[2]; };

// CDNA5 16-bit A fragment: lane = M + 16*h; VGPR j<4: K = h*8+2j+p ; j>=4: K = 16+h*8+2(j-4)+p
// a_swz(m, kk) -> linear index into a [32 lanes][16 elem] LDS image of the fragment.
static __device__ __forceinline__ int a_swz(int m, int kk) {
  int hi = (kk >= 16) ? 1 : 0;
  int k2 = kk & 15;
  int hh = k2 >> 3;
  int j  = ((k2 & 7) >> 1) + hi * 4;
  int e  = 2 * j + (kk & 1);
  return (m + 16 * hh) * 16 + e;
}

#if HAVE_TDM
// Issue a TDM 2D tile load (bf16, rows x cols, row stride in elements) into LDS.
static __device__ __forceinline__ void tdm_load_tile_bf16(
    unsigned lds_off, const unsigned short* gptr, int rows, int cols,
    long rowStrideElems, unsigned remCols, unsigned remRows)
{
  unsigned long long ga = (unsigned long long)(uintptr_t)gptr;
  u32x4 g0;
  g0[0] = 1u;                                           // count=1 (valid), user mode
  g0[1] = lds_off;                                      // LDS byte address
  g0[2] = (unsigned)(ga & 0xFFFFFFFFull);               // global_addr[31:0]
  g0[3] = (unsigned)((ga >> 32) & 0x1FFFFFFull) | (2u << 30);  // addr[56:32] | type=2
  unsigned long long st0 = (unsigned long long)rowStrideElems;
  i32x8 g1;
  g1[0] = (int)(1u << 16);                              // data_size=1 (2 bytes/elem)
  g1[1] = (int)((remCols & 0xFFFFu) << 16);             // tensor_dim0[15:0]
  g1[2] = (int)(((remCols >> 16) & 0xFFFFu) | ((remRows & 0xFFFFu) << 16));
  g1[3] = (int)(((remRows >> 16) & 0xFFFFu) | ((unsigned)cols << 16)); // tile_dim0
  g1[4] = (int)((unsigned)rows);                        // tile_dim1 (tile_dim2=0)
  g1[5] = (int)(st0 & 0xFFFFFFFFull);                   // tensor_dim0_stride[31:0]
  g1[6] = (int)((st0 >> 32) & 0xFFFFull);               // stride[47:32] | dim1_stride=0
  g1[7] = 0;
  i32x4 z4 = {0, 0, 0, 0};
#if __clang_major__ >= 23
  i32x8 z8 = {0, 0, 0, 0, 0, 0, 0, 0};
  __builtin_amdgcn_tensor_load_to_lds(g0, g1, z4, z4, z8, 0);
#else
  __builtin_amdgcn_tensor_load_to_lds(g0, g1, z4, z4, 0);
#endif
}
#endif

// ======================= WMMA gather-GEMM ======================
// out[n, co] = sum_k sum_ci X[idx[n,k], ci] * Wb[k*Ci+ci, co] (+bias)
// idx == nullptr -> identity gather (1x1 conv / linear; K==1).
// Block = 128 threads = 4 waves. Tile = 32 rows x 64 cols:
//   - one gathered A tile (two 16x32 fragments) shared by all 4 waves
//   - each wave owns a 16-wide co strip; its B fragment feeds TWO WMMAs
__global__ void __launch_bounds__(128)
gather_conv_wmma(const float* __restrict__ X, const int* __restrict__ idx,
                 const unsigned short* __restrict__ Wb, const float* __restrict__ bias,
                 int N, int K, int Ci, int Co, float* __restrict__ out)
{
  __shared__ __align__(16) unsigned short As[2][16 * 32]; // fragment-swizzled A images

  const int tid  = threadIdx.x;
  const int wave = tid >> 5;
  const int lane = tid & 31;
  const int row0 = blockIdx.x * 32;
  const int co0  = blockIdx.y * 64 + wave * 16;
  const int arow = lane & 15;
  const int h    = lane >> 4;
  const int r    = tid >> 2;        // staging row 0..31
  const int sub  = tid & 3;         // 8-col group within the 32-wide ci chunk
  const bool vec4 = ((Ci & 31) == 0);   // full chunks -> safe b128 loads

  v8f acc0 = {};
  v8f acc1 = {};

  for (int k = 0; k < K; ++k) {
    int grow = row0 + r;
    int site = -1;
    if (grow < N) site = idx ? idx[grow * K + k] : grow;
    if (idx && grow < N && (k + 1) < K) {               // prefetch next rulebook row
      int ns = idx[grow * K + k + 1];
      if (ns >= 0) __builtin_prefetch(&X[(long)ns * Ci], 0, 0);
    }
    unsigned short* img = As[r >> 4];
    const int rm = r & 15;
    for (int cb = 0; cb < Ci; cb += 32) {
      // ---- stage A (gather + f32->bf16, written in fragment order) ----
      const int c0 = cb + sub * 8;
      if (vec4) {
        v4f va = {0.f, 0.f, 0.f, 0.f}, vb = {0.f, 0.f, 0.f, 0.f};
        if (site >= 0) {
          const float* xr = &X[(long)site * Ci + c0];
          va = *(const v4f*)xr;
          vb = *(const v4f*)(xr + 4);
        }
        #pragma unroll
        for (int u = 0; u < 4; ++u) {
          img[a_swz(rm, sub * 8 + u)]     = f32_bf16(va[u]);
          img[a_swz(rm, sub * 8 + 4 + u)] = f32_bf16(vb[u]);
        }
      } else {
        #pragma unroll
        for (int u = 0; u < 8; ++u) {
          int c = c0 + u;
          float v = (site >= 0 && c < Ci) ? X[(long)site * Ci + c] : 0.f;
          img[a_swz(rm, sub * 8 + u)] = f32_bf16(v);
        }
      }
      __syncthreads();
      // ---- fragments: A via b128 LDS loads; B direct bf16 global (shared) ----
      BF16Frag a0, a1, b;
      const u32x4* ap0 = (const u32x4*)As[0];
      const u32x4* ap1 = (const u32x4*)As[1];
      a0.q[0] = ap0[lane * 2];
      a0.q[1] = ap0[lane * 2 + 1];
      a1.q[0] = ap1[lane * 2];
      a1.q[1] = ap1[lane * 2 + 1];
      const int co = co0 + arow;
      #pragma unroll
      for (int e = 0; e < 16; ++e) {
        int ci = cb + h * 8 + e + (e & 8);
        b.u[e] = (ci < Ci && co < Co) ? Wb[((long)k * Ci + ci) * Co + co]
                                      : (unsigned short)0;
      }
      acc0 = __builtin_amdgcn_wmma_f32_16x16x32_bf16(false, a0.v, false, b.v,
                                                     (short)0, acc0, false, false);
      acc1 = __builtin_amdgcn_wmma_f32_16x16x32_bf16(false, a1.v, false, b.v,
                                                     (short)0, acc1, false, false);
      __syncthreads();
    }
  }
  // ---- store D: VGPR d -> row = d + 8*h, col = lane&15 ----
  const int co = co0 + arow;
  if (co < Co) {
    const float bv = bias ? bias[co] : 0.f;
    #pragma unroll
    for (int d = 0; d < 8; ++d) {
      int g0r = row0 + d + 8 * h;
      int g1r = row0 + 16 + d + 8 * h;
      if (g0r < N) out[(long)g0r * Co + co] = acc0[d] + bv;
      if (g1r < N) out[(long)g1r * Co + co] = acc1[d] + bv;
    }
  }
}

// ======================= WMMA deconv (scatter) ======================
// contrib = Y(Nc x Cb) @ Wb[k](Cb x Ca); out[idx[n,k]] += contrib[n]
// Cb, Ca are multiples of 32/16 here, so the TDM weight tile never goes OOB.
__global__ void __launch_bounds__(32)
deconv_scatter_wmma(const float* __restrict__ Y, const int* __restrict__ idx,
                    const unsigned short* __restrict__ Wb, int Nc, int Cb, int Ca,
                    float* __restrict__ out)
{
  __shared__ __align__(16) unsigned short As[16 * 32];   // fragment-swizzled
  __shared__ __align__(16) unsigned short Bs[32 * 16];   // raw row-major tile
  const int lane = threadIdx.x;
  const int row0 = blockIdx.x * 16;
  const int co0  = blockIdx.y * 16;
  const int k    = blockIdx.z;                           // 0..7
  const int r    = lane >> 1;
  const int part = lane & 1;
  const int arow = lane & 15;
  const int h    = lane >> 4;

  v8f acc = {};
  for (int cb = 0; cb < Cb; cb += 32) {
    // ---- stage A ----
    int grow = row0 + r;
    #pragma unroll
    for (int t = 0; t < 16; ++t) {
      int kk = part * 16 + t;
      int c  = cb + kk;
      float v = (grow < Nc && c < Cb) ? Y[(long)grow * Cb + c] : 0.f;
      As[a_swz(r, kk)] = f32_bf16(v);
    }
    // ---- stage B: TDM async tile load (fallback: manual) ----
#if HAVE_TDM
    tdm_load_tile_bf16((unsigned)(uintptr_t)&Bs[0],
                       Wb + ((long)k * Cb + cb) * Ca + co0,
                       /*rows*/32, /*cols*/16, /*rowStride*/(long)Ca,
                       (unsigned)(Ca - co0), (unsigned)(Cb - cb));
#else
    #pragma unroll
    for (int t = 0; t < 16; ++t)
      Bs[lane * 16 + t] = Wb[((long)k * Cb + cb + lane) * Ca + co0 + t];
#endif
    __syncthreads();
#if HAVE_TDM
    __builtin_amdgcn_s_wait_tensorcnt(0);
#endif
    BF16Frag a, b;
    const u32x4* ap = (const u32x4*)As;
    a.q[0] = ap[lane * 2];
    a.q[1] = ap[lane * 2 + 1];
    #pragma unroll
    for (int e = 0; e < 16; ++e)
      b.u[e] = Bs[(h * 8 + e + (e & 8)) * 16 + arow];
    acc = __builtin_amdgcn_wmma_f32_16x16x32_bf16(false, a.v, false, b.v,
                                                  (short)0, acc, false, false);
    __syncthreads();
  }
  #pragma unroll
  for (int d = 0; d < 8; ++d) {
    int grow = row0 + d + 8 * h;
    int co   = co0 + arow;
    if (grow < Nc && co < Ca) {
      int fi = idx[grow * 8 + k];
      if (fi >= 0) atomicAdd(&out[(long)fi * Ca + co], acc[d]);
    }
  }
}

// ======================= elementwise / reduction ======================
__global__ void memzero_k(float* p, long n) {
  long i = (long)blockIdx.x * blockDim.x + threadIdx.x;
  if (i < n) p[i] = 0.f;
}
__global__ void copy_k(float* d, const float* s, long n) {
  long i = (long)blockIdx.x * blockDim.x + threadIdx.x;
  if (i < n) d[i] = s[i];
}
__global__ void add_k(float* d, const float* s, long n) {
  long i = (long)blockIdx.x * blockDim.x + threadIdx.x;
  if (i < n) d[i] += s[i];
}
__global__ void f32_to_bf16_k(const float* __restrict__ s, unsigned short* __restrict__ d, long n) {
  long i = (long)blockIdx.x * blockDim.x + threadIdx.x;
  if (i < n) d[i] = f32_bf16(s[i]);
}
__global__ void copycols_k(float* dst, int dstride, int dcol0,
                           const float* src, int sstride, int ncols, int N) {
  long i = (long)blockIdx.x * blockDim.x + threadIdx.x;
  if (i < (long)N * ncols) {
    int r = (int)(i / ncols), c = (int)(i % ncols);
    dst[(long)r * dstride + dcol0 + c] = src[(long)r * sstride + c];
  }
}
__global__ void colstats_k(const float* __restrict__ x, int N, int C,
                           float* __restrict__ stats) {
  extern __shared__ float sm[];            // [0,C)=sum, [C,2C)=sumsq
  for (int i = threadIdx.x; i < 2 * C; i += blockDim.x) sm[i] = 0.f;
  __syncthreads();
  int r = blockIdx.x * blockDim.x + threadIdx.x;
  if (r < N) {
    const float* row = x + (long)r * C;
    for (int c = 0; c < C; ++c) {
      float v = row[c];
      atomicAdd(&sm[c], v);                // ds_add_f32
      atomicAdd(&sm[C + c], v * v);
    }
  }
  __syncthreads();
  for (int i = threadIdx.x; i < 2 * C; i += blockDim.x)
    atomicAdd(&stats[i], sm[i]);           // global_atomic_add_f32
}
__global__ void bnapply_k(const float* __restrict__ x, const float* __restrict__ stats,
                          const float* __restrict__ g, const float* __restrict__ b,
                          int N, int C, float* __restrict__ y) {
  long i = (long)blockIdx.x * blockDim.x + threadIdx.x;
  if (i < (long)N * C) {
    int c = (int)(i % C);
    float m = stats[c] / (float)N;
    float v = stats[C + c] / (float)N - m * m;
    float z = (x[i] - m) * rsqrtf(v + BN_EPS) * g[c] + b[c];
    y[i] = fmaxf(z, 0.f);
  }
}

// ======================= host-side driver ======================
static const int NS_[7]     = {50000, 20000, 8000, 3200, 1280, 640, 320};
static const int PLANES_[7] = {32, 64, 96, 128, 160, 192, 224};

struct BN  { const float* g; const float* b; };
struct Blk {
  BN bn1; const float* w1; BN bn2; const float* w2; const float* nin;
  const unsigned short* w1b; const unsigned short* w2b; const unsigned short* ninb;
};
struct Lvl {
  Blk pre[2]; BN bn_down; const float* w_down; BN bn_up; const float* w_up;
  Blk post[2]; bool has_inner;
  const unsigned short* w_downb; const unsigned short* w_upb;
};

struct Cursor {
  void* const* in; int i;
  const float* f() { return (const float*)in[i++]; }
  const int*   d() { return (const int*)in[i++]; }
};

static void readBlk(Cursor& c, Blk& b, bool nin) {
  b.bn1.g = c.f(); b.bn1.b = c.f(); b.w1 = c.f();
  b.bn2.g = c.f(); b.bn2.b = c.f(); b.w2 = c.f();
  b.nin = nin ? c.f() : nullptr;
  b.w1b = b.w2b = b.ninb = nullptr;
}
static void readLvl(Cursor& c, Lvl* L, int lvl) {
  readBlk(c, L[lvl].pre[0], false);
  readBlk(c, L[lvl].pre[1], false);
  L[lvl].has_inner = (lvl + 1 < 7);
  if (L[lvl].has_inner) {
    L[lvl].bn_down.g = c.f(); L[lvl].bn_down.b = c.f(); L[lvl].w_down = c.f();
    readLvl(c, L, lvl + 1);                               // "inner" before bn_up
    L[lvl].bn_up.g = c.f(); L[lvl].bn_up.b = c.f(); L[lvl].w_up = c.f();
    readBlk(c, L[lvl].post[0], true);
    readBlk(c, L[lvl].post[1], false);
  }
}

struct Ctx {
  char* ws; size_t off; hipStream_t s; float* stats;
  void* allocB(size_t bytes) {
    void* p = (void*)(ws + off);
    off += ((bytes + 255) & ~(size_t)255);
    return p;
  }
  float* alloc(long n) { return (float*)allocB((size_t)n * 4); }
};

static inline int g1(long n) { return (int)((n + 255) / 256); }

static void zeroBuf(Ctx& c, float* p, long n) { memzero_k<<<g1(n), 256, 0, c.s>>>(p, n); }
static void copyBuf(Ctx& c, float* d, const float* s, long n) { copy_k<<<g1(n), 256, 0, c.s>>>(d, s, n); }
static void addBuf(Ctx& c, float* d, const float* s, long n) { add_k<<<g1(n), 256, 0, c.s>>>(d, s, n); }

static const unsigned short* cvtW(Ctx& c, const float* w, long n) {
  unsigned short* d = (unsigned short*)c.allocB((size_t)n * 2);
  f32_to_bf16_k<<<g1(n), 256, 0, c.s>>>(w, d, n);
  return d;
}
static void cvtLvl(Ctx& c, Lvl* L, int lvl) {
  const long a = PLANES_[lvl];
  L[lvl].pre[0].w1b = cvtW(c, L[lvl].pre[0].w1, 27 * a * a);
  L[lvl].pre[0].w2b = cvtW(c, L[lvl].pre[0].w2, 27 * a * a);
  L[lvl].pre[1].w1b = cvtW(c, L[lvl].pre[1].w1, 27 * a * a);
  L[lvl].pre[1].w2b = cvtW(c, L[lvl].pre[1].w2, 27 * a * a);
  if (!L[lvl].has_inner) return;
  const long b = PLANES_[lvl + 1];
  L[lvl].w_downb = cvtW(c, L[lvl].w_down, 8 * a * b);
  cvtLvl(c, L, lvl + 1);
  L[lvl].w_upb   = cvtW(c, L[lvl].w_up, 8 * b * a);
  L[lvl].post[0].ninb = cvtW(c, L[lvl].post[0].nin, 2 * a * a);
  L[lvl].post[0].w1b  = cvtW(c, L[lvl].post[0].w1, 27 * 2 * a * a);
  L[lvl].post[0].w2b  = cvtW(c, L[lvl].post[0].w2, 27 * a * a);
  L[lvl].post[1].w1b  = cvtW(c, L[lvl].post[1].w1, 27 * a * a);
  L[lvl].post[1].w2b  = cvtW(c, L[lvl].post[1].w2, 27 * a * a);
}

static void bnrelu(Ctx& c, const float* x, BN bn, int N, int C, float* y) {
  zeroBuf(c, c.stats, 2 * C);
  colstats_k<<<g1(N), 256, 2 * C * sizeof(float), c.s>>>(x, N, C, c.stats);
  bnapply_k<<<g1((long)N * C), 256, 0, c.s>>>(x, c.stats, bn.g, bn.b, N, C, y);
}
static void gconv(Ctx& c, const float* x, const int* idx, int N, int K,
                  int Ci, int Co, const unsigned short* wb, const float* bias, float* out) {
  dim3 g((N + 31) / 32, (Co + 63) / 64);
  gather_conv_wmma<<<g, 128, 0, c.s>>>(x, idx, wb, bias, N, K, Ci, Co, out);
}

static void runBlock(Ctx& c, const float* x, int N, int Ci, int Co,
                     const int* idx, const Blk& B, float* out,
                     float* tY /*>=N*Ci*/, float* tZ /*>=N*Co*/) {
  if (B.nin) gconv(c, x, nullptr, N, 1, Ci, Co, B.ninb, nullptr, out);
  else       copyBuf(c, out, x, (long)N * Co);
  bnrelu(c, x, B.bn1, N, Ci, tY);
  gconv(c, tY, idx, N, 27, Ci, Co, B.w1b, nullptr, tZ);
  bnrelu(c, tZ, B.bn2, N, Co, tY);
  gconv(c, tY, idx, N, 27, Co, Co, B.w2b, nullptr, tZ);
  addBuf(c, out, tZ, (long)N * Co);
}

static float* unetRun(Ctx& c, int lvl, const float* x, const Lvl* L,
                      const int* const* idxs, const int* const* downs) {
  const int N = NS_[lvl], a = PLANES_[lvl];
  float* tA = c.alloc((long)N * 2 * a);
  float* tB = c.alloc((long)N * 2 * a);
  float* b0 = c.alloc((long)N * a);
  runBlock(c, x, N, a, a, idxs[lvl], L[lvl].pre[0], b0, tA, tB);
  float* b1 = c.alloc((long)N * a);
  runBlock(c, b0, N, a, a, idxs[lvl], L[lvl].pre[1], b1, tA, tB);
  if (!L[lvl].has_inner) return b1;

  const int Nn = NS_[lvl + 1], bch = PLANES_[lvl + 1];
  bnrelu(c, b1, L[lvl].bn_down, N, a, tA);
  float* down = c.alloc((long)Nn * bch);
  gconv(c, tA, downs[lvl], Nn, 8, a, bch, L[lvl].w_downb, nullptr, down);
  float* innerOut = unetRun(c, lvl + 1, down, L, idxs, downs);
  bnrelu(c, innerOut, L[lvl].bn_up, Nn, bch, tA);      // tA >= Nn*bch
  float* up = c.alloc((long)N * a);
  zeroBuf(c, up, (long)N * a);
  {
    dim3 g((Nn + 15) / 16, (a + 15) / 16, 8);
    deconv_scatter_wmma<<<g, 32, 0, c.s>>>(tA, downs[lvl], L[lvl].w_upb, Nn, bch, a, up);
  }
  float* cat = c.alloc((long)N * 2 * a);
  copycols_k<<<g1((long)N * a), 256, 0, c.s>>>(cat, 2 * a, 0, b1, a, a, N);
  copycols_k<<<g1((long)N * a), 256, 0, c.s>>>(cat, 2 * a, a, up, a, a, N);
  float* p0 = c.alloc((long)N * a);
  runBlock(c, cat, N, 2 * a, a, idxs[lvl], L[lvl].post[0], p0, tA, tB);
  float* p1 = c.alloc((long)N * a);
  runBlock(c, p0, N, a, a, idxs[lvl], L[lvl].post[1], p1, tA, tB);
  return p1;
}

extern "C" void kernel_launch(void* const* d_in, const int* in_sizes, int n_in,
                              void* d_out, int out_size, void* d_ws, size_t ws_size,
                              hipStream_t stream) {
  (void)in_sizes; (void)n_in; (void)out_size; (void)ws_size;
  Cursor cur{d_in, 0};
  const float* features = cur.f();                  // (50000, 3)
  const float* w_in     = cur.f();                  // (27, 3, 32)
  Lvl L[7];
  readLvl(cur, L, 0);
  BN bn_out; bn_out.g = cur.f(); bn_out.b = cur.f();
  const float* w_lin = cur.f();                     // (32, 20)
  const float* b_lin = cur.f();                     // (20,)
  const int* idxs[7];  for (int l = 0; l < 7; ++l) idxs[l]  = cur.d();
  const int* downs[6]; for (int l = 0; l < 6; ++l) downs[l] = cur.d();

  Ctx c; c.ws = (char*)d_ws; c.off = 0; c.s = stream;
  c.stats = c.alloc(2 * 448);

  // convert all weights to bf16 (one pass; tiny vs. activation traffic)
  const unsigned short* w_inb  = cvtW(c, w_in, 27L * 3 * 32);
  cvtLvl(c, L, 0);
  const unsigned short* w_linb = cvtW(c, w_lin, 32L * 20);

  // input conv: 3 -> 32 (Ci=3 zero-padded inside the 32-wide chunk)
  float* x0 = c.alloc((long)NS_[0] * PLANES_[0]);
  gconv(c, features, idxs[0], NS_[0], 27, 3, PLANES_[0], w_inb, nullptr, x0);

  float* u = unetRun(c, 0, x0, L, idxs, downs);

  float* t = c.alloc((long)NS_[0] * PLANES_[0]);
  bnrelu(c, u, bn_out, NS_[0], PLANES_[0], t);
  gconv(c, t, nullptr, NS_[0], 1, PLANES_[0], 20, w_linb, b_lin, (float*)d_out);
}
